// EdgeAwareGNN_4466765988228
// MI455X (gfx1250) — matrix operand order repaired and checked by reading, not compiled, not run on hardware
//
#include <hip/hip_runtime.h>
#include <hip/hip_bf16.h>
#include <stdint.h>

// ---------------------------------------------------------------------------
// EdgeAwareGNN for MI455X (gfx1250, wave32, WMMA, async-LDS staging).
//
// msg = (h_mlp[e] outer hsrc[e]) @ Bpacked as a fused bf16 WMMA GEMM.
// A-fragments are generated in registers: per 32-wide K chunk the outer
// product collapses to (broadcast scalar) x (cached packed-bf16 hsrc
// fragment), computed with native v_pk_mul_bf16. Bias rows ride in the GEMM
// via +1.0 padding of the staged LDS rows (branch-free indexing).
// Staging uses global_load_async_to_lds_b128 (ASYNCcnt).
// ---------------------------------------------------------------------------

#define N_NODES 32768
#define N_EDGES 65536
#define HIDDEN  64

typedef __attribute__((ext_vector_type(16))) __bf16          v16bf;
typedef __attribute__((ext_vector_type(16))) unsigned short  v16u;
typedef __attribute__((ext_vector_type(8)))  unsigned int    v8u;
typedef __attribute__((ext_vector_type(8)))  float           v8f;

union BF16x16 { v16bf bf; v16u us; v8u u32; };

__device__ __forceinline__ unsigned short f2bf(float f) {
    unsigned int u = __float_as_uint(f);
    u += 0x7FFFu + ((u >> 16) & 1u);          // round-to-nearest-even
    return (unsigned short)(u >> 16);
}
__device__ __forceinline__ unsigned pack2(float lo, float hi) {
    return (unsigned)f2bf(lo) | ((unsigned)f2bf(hi) << 16);
}
__device__ __forceinline__ unsigned bcast_bf(float f) {
    unsigned u = (unsigned)f2bf(f);
    return u | (u << 16);
}
// native VOP3P packed bf16 multiply (CDNA5 VOP3P opcode 42)
__device__ __forceinline__ unsigned pk_mul_bf16(unsigned a, unsigned b) {
    unsigned d;
    asm("v_pk_mul_bf16 %0, %1, %2" : "=v"(d) : "v"(a), "v"(b));
    return d;
}
// async global->LDS 128b copy (tracked by ASYNCcnt)
__device__ __forceinline__ void async_g2l_b128(void* lds_ptr, const void* gptr) {
    unsigned loff = (unsigned)(uintptr_t)lds_ptr;            // low 32b = LDS offset
    unsigned long long ga = (unsigned long long)(uintptr_t)gptr;
    asm volatile("global_load_async_to_lds_b128 %0, %1, off"
                 :: "v"(loff), "v"(ga) : "memory");
}
__device__ __forceinline__ void wait_async0() {
    asm volatile("s_wait_asynccnt 0x0" ::: "memory");
}

// ---------------------------------------------------------------------------
// small utility kernels
// ---------------------------------------------------------------------------
__global__ void zero_kernel(float* p, int n) {
    int i = blockIdx.x * 256 + threadIdx.x;
    if (i < n) p[i] = 0.0f;
}

__global__ void count_kernel(const int* dst, float* cnt) {
    int e = blockIdx.x * 256 + threadIdx.x;
    if (e < N_EDGES) unsafeAtomicAdd(&cnt[dst[e]], 1.0f);
}

// per-edge first MLP layers: h1e = relu(ea@e1_w1+b), h2e = relu(ea@e2_w1+b)
__global__ __launch_bounds__(128) void edge_mlp_kernel(
    const float* __restrict__ ea,
    const float* __restrict__ w1a, const float* __restrict__ b1a,   // [4,64],[64]
    const float* __restrict__ w1b, const float* __restrict__ b1b,   // [4,128],[128]
    float* __restrict__ h1e, float* __restrict__ h2e) {
    int e = blockIdx.x, t = threadIdx.x;
    __shared__ float a[4];
    if (t < 4) a[t] = ea[e * 4 + t];
    __syncthreads();
    if (t < 64) {
        float v = b1a[t];
        #pragma unroll
        for (int i = 0; i < 4; ++i) v += a[i] * w1a[i * 64 + t];
        h1e[(size_t)e * 64 + t] = fmaxf(v, 0.0f);
    }
    {
        float v = b1b[t];
        #pragma unroll
        for (int i = 0; i < 4; ++i) v += a[i] * w1b[i * 128 + t];
        h2e[(size_t)e * 128 + t] = fmaxf(v, 0.0f);
    }
}

// Pack B[kf,o] (kf = k*IN + i over w2 rows, then IN bias rows, then zero pad)
// into bf16 tiles in the WMMA-B fragment layout (ISA 7.12.2):
//   addr = ((c*4 + t)*32 + lane)*16 + j ;  K = c*32 + (j/8)*16 + (lane/16)*8 + j%8,
//   N = t*16 + lane%16
__global__ void prep_b_kernel(unsigned short* __restrict__ out,
                              const float* __restrict__ w2,
                              const float* __restrict__ b2,
                              int KH, int inLog2, int chunks) {
    int idx = blockIdx.x * 256 + threadIdx.x;
    int total = chunks * 4 * 32 * 16;
    if (idx >= total) return;
    int j    = idx & 15;
    int lane = (idx >> 4) & 31;
    int ct   = idx >> 9;
    int t    = ct & 3;
    int c    = ct >> 2;
    int kf   = c * 32 + (j >> 3) * 16 + (lane >> 4) * 8 + (j & 7);
    int o    = t * 16 + (lane & 15);
    int IN   = 1 << inLog2;
    int KTOT = KH << inLog2;
    float v = 0.0f;
    if (kf < KTOT) {
        int k = kf >> inLog2, i = kf & (IN - 1);
        v = w2[((size_t)(k << inLog2)) * 64 + (size_t)i * 64 + o];
    } else if (kf < KTOT + IN) {
        v = b2[(size_t)(kf - KTOT) * 64 + o];
    }
    out[idx] = f2bf(v);
}

// ---------------------------------------------------------------------------
// Layer-1 message kernel: K = 64*4 (+4 bias, pad to 288 = 9 chunks).
// Staged h1e rows padded to 72: [64]=1.0 (bias), [65..71]=0 (pad) ->
// uniform a = H[kf>>2] * X[kf&3], no branches.
// ---------------------------------------------------------------------------
__global__ __launch_bounds__(128) void msg1_kernel(
    const float* __restrict__ x, const float* __restrict__ h1e,
    const int* __restrict__ src, const int* __restrict__ dst,
    const unsigned short* __restrict__ B1p, float* __restrict__ nodeacc) {
    __shared__ float xs[16 * 4];
    __shared__ float he[16 * 72];
    __shared__ int   ssrc[16], sdst[16];
    int tid = threadIdx.x, eb = blockIdx.x * 16;
    if (tid < 16) { ssrc[tid] = src[eb + tid]; sdst[tid] = dst[eb + tid]; }
    {   // pad columns 64..71 of each row: 1.0 then zeros
        int e = tid >> 3, w = tid & 7;
        he[e * 72 + 64 + w] = (w == 0) ? 1.0f : 0.0f;
    }
    __syncthreads();
    if (tid < 64) xs[tid] = x[(size_t)ssrc[tid >> 2] * 4 + (tid & 3)];
    // async-stage 16 rows x 64 floats (16B granules)
    for (int q = tid; q < 256; q += 128) {
        int e = q >> 4, w = (q & 15) * 4;
        async_g2l_b128(&he[e * 72 + w], h1e + (size_t)(eb + e) * 64 + w);
    }
    wait_async0();
    __syncthreads();

    int lane = tid & 31, wv = tid >> 5, m = lane & 15, g = lane >> 4;
    unsigned xp01 = pack2(xs[m * 4 + 0], xs[m * 4 + 1]);
    unsigned xp23 = pack2(xs[m * 4 + 2], xs[m * 4 + 3]);
    const float* H = &he[m * 72];

    v8f acc = {};
    #pragma unroll
    for (int c = 0; c < 9; ++c) {
        int kb = 8 * c + 2 * g;
        unsigned k0 = bcast_bf(H[kb]);
        unsigned k1 = bcast_bf(H[kb + 1]);
        unsigned k2 = bcast_bf(H[kb + 4]);
        unsigned k3 = bcast_bf(H[kb + 5]);
        BF16x16 av, bv;
        av.u32[0] = pk_mul_bf16(k0, xp01); av.u32[1] = pk_mul_bf16(k0, xp23);
        av.u32[2] = pk_mul_bf16(k1, xp01); av.u32[3] = pk_mul_bf16(k1, xp23);
        av.u32[4] = pk_mul_bf16(k2, xp01); av.u32[5] = pk_mul_bf16(k2, xp23);
        av.u32[6] = pk_mul_bf16(k3, xp01); av.u32[7] = pk_mul_bf16(k3, xp23);
        bv.us = *(const v16u*)(B1p + ((size_t)(c * 4 + wv) * 32 + lane) * 16);
        acc = __builtin_amdgcn_wmma_f32_16x16x32_bf16(
                  false, av.bf, false, bv.bf, (short)0, acc, false, false);
    }
    int n = lane & 15;
    #pragma unroll
    for (int r = 0; r < 8; ++r) {
        int d = sdst[r + 8 * g];                     // D: M = r + 8*g
        unsafeAtomicAdd(&nodeacc[(size_t)d * 64 + wv * 16 + n], acc[r]);
    }
}

// ---------------------------------------------------------------------------
// Layer-2 message kernel: K = 128*64 (+64 bias) = 258 chunks = 129 kc x 2.
// Each chunk has constant k -> A = scalar * cached packed-bf16 hsrc fragment.
// h2e rows staged 132 wide with [128]=1.0 so bias chunks need no branch.
// ---------------------------------------------------------------------------
__global__ __launch_bounds__(128) void msg2_kernel(
    const float* __restrict__ h1, const float* __restrict__ h2e,
    const int* __restrict__ src, const int* __restrict__ dst,
    const unsigned short* __restrict__ B2p, float* __restrict__ nodeacc) {
    __shared__ float hs[16 * 64];
    __shared__ float he[16 * 132];
    __shared__ int   sdst[16];
    int tid = threadIdx.x, eb = blockIdx.x * 16;
    if (tid < 16) sdst[tid] = dst[eb + tid];
    if (tid < 64) {                                   // pad cols 128..131
        int e = tid >> 2, w = tid & 3;
        he[e * 132 + 128 + w] = (w == 0) ? 1.0f : 0.0f;
    }
    // async-stage h2e rows (16 x 128 floats, contiguous per edge)
    for (int q = tid; q < 512; q += 128) {
        int e = q >> 5, w = (q & 31) * 4;
        async_g2l_b128(&he[e * 132 + w], h2e + (size_t)(eb + e) * 128 + w);
    }
    // async-gather hsrc rows (16 x 64 floats from scattered nodes)
    for (int q = tid; q < 256; q += 128) {
        int e = q >> 4, w = (q & 15) * 4;
        int sn = src[eb + e];
        async_g2l_b128(&hs[e * 64 + w], h1 + (size_t)sn * 64 + w);
    }
    wait_async0();
    __syncthreads();

    int lane = tid & 31, wv = tid >> 5, m = lane & 15, g = lane >> 4;

    // packed-bf16 hsrc fragments for ibase = 0 / 32, reused across all kc
    unsigned hp0[8], hp1[8];
    #pragma unroll
    for (int p = 0; p < 8; ++p) {
        int kk = (p < 4) ? (8 * g + 2 * p) : (16 + 8 * g + 2 * p - 8);
        hp0[p] = pack2(hs[m * 64 + kk],      hs[m * 64 + kk + 1]);
        hp1[p] = pack2(hs[m * 64 + 32 + kk], hs[m * 64 + 32 + kk + 1]);
    }

    v8f acc = {};
    const float* herow = &he[m * 132];
    for (int kc = 0; kc < 129; ++kc) {                // kc==128 -> bias (H=1.0)
        unsigned sb = bcast_bf(herow[kc]);
        BF16x16 a0, a1, b0, b1;
        #pragma unroll
        for (int p = 0; p < 8; ++p) a0.u32[p] = pk_mul_bf16(sb, hp0[p]);
        #pragma unroll
        for (int p = 0; p < 8; ++p) a1.u32[p] = pk_mul_bf16(sb, hp1[p]);
        int c0 = kc * 2, c1 = c0 + 1;
        b0.us = *(const v16u*)(B2p + ((size_t)(c0 * 4 + wv) * 32 + lane) * 16);
        b1.us = *(const v16u*)(B2p + ((size_t)(c1 * 4 + wv) * 32 + lane) * 16);
        acc = __builtin_amdgcn_wmma_f32_16x16x32_bf16(
                  false, a0.bf, false, b0.bf, (short)0, acc, false, false);
        acc = __builtin_amdgcn_wmma_f32_16x16x32_bf16(
                  false, a1.bf, false, b1.bf, (short)0, acc, false, false);
    }
    int n = lane & 15;
    #pragma unroll
    for (int r = 0; r < 8; ++r) {
        int d = sdst[r + 8 * g];
        unsafeAtomicAdd(&nodeacc[(size_t)d * 64 + wv * 16 + n], acc[r]);
    }
}

// ---------------------------------------------------------------------------
// node update: agg/mean + x@root + bias -> relu -> layernorm.  64 thr / node
// ---------------------------------------------------------------------------
__global__ __launch_bounds__(64) void node_update_kernel(
    const float* __restrict__ acc, const float* __restrict__ cnt,
    const float* __restrict__ infeat, int indim,
    const float* __restrict__ root, const float* __restrict__ bias,
    const float* __restrict__ lng, const float* __restrict__ lnb,
    float* __restrict__ out) {
    int n = blockIdx.x, o = threadIdx.x;
    __shared__ float sin_[64];
    __shared__ float sv[64];
    if (o < indim) sin_[o] = infeat[(size_t)n * indim + o];
    __syncthreads();
    float v = acc[(size_t)n * 64 + o] / fmaxf(cnt[n], 1.0f);
    for (int k = 0; k < indim; ++k) v += sin_[k] * root[k * 64 + o];
    v += bias[o];
    v = fmaxf(v, 0.0f);
    sv[o] = v;
    __syncthreads();
    float mu = 0.0f;
    for (int k = 0; k < 64; ++k) mu += sv[k];
    mu *= (1.0f / 64.0f);
    float var = 0.0f;
    for (int k = 0; k < 64; ++k) { float d = sv[k] - mu; var += d * d; }
    var *= (1.0f / 64.0f);
    out[(size_t)n * 64 + o] = (v - mu) * rsqrtf(var + 1e-5f) * lng[o] + lnb[o];
}

// global mean pool: one block per output channel
__global__ __launch_bounds__(256) void pool_kernel(const float* __restrict__ h,
                                                   float* __restrict__ out) {
    int o = blockIdx.x, t = threadIdx.x;
    float s = 0.0f;
    for (int n = t; n < N_NODES; n += 256) s += h[(size_t)n * 64 + o];
    __shared__ float red[256];
    red[t] = s;
    __syncthreads();
    for (int w = 128; w > 0; w >>= 1) {
        if (t < w) red[t] += red[t + w];
        __syncthreads();
    }
    if (t == 0) out[o] = red[0] * (1.0f / (float)N_NODES);
}

// ---------------------------------------------------------------------------
extern "C" void kernel_launch(void* const* d_in, const int* in_sizes, int n_in,
                              void* d_out, int out_size, void* d_ws, size_t ws_size,
                              hipStream_t stream) {
    (void)in_sizes; (void)n_in; (void)out_size; (void)ws_size;
    const float* x     = (const float*)d_in[0];
    const float* ea    = (const float*)d_in[1];
    const int*   eidx  = (const int*)d_in[2];      // [2, E]
    const float* e1w1  = (const float*)d_in[4];
    const float* e1b1  = (const float*)d_in[5];
    const float* e1w2  = (const float*)d_in[6];
    const float* e1b2  = (const float*)d_in[7];
    const float* root1 = (const float*)d_in[8];
    const float* bias1 = (const float*)d_in[9];
    const float* ln1g  = (const float*)d_in[10];
    const float* ln1b  = (const float*)d_in[11];
    const float* e2w1  = (const float*)d_in[12];
    const float* e2b1  = (const float*)d_in[13];
    const float* e2w2  = (const float*)d_in[14];
    const float* e2b2  = (const float*)d_in[15];
    const float* root2 = (const float*)d_in[16];
    const float* bias2 = (const float*)d_in[17];
    const float* ln2g  = (const float*)d_in[18];
    const float* ln2b  = (const float*)d_in[19];
    const int* src = eidx;
    const int* dst = eidx + N_EDGES;

    char* ws = (char*)d_ws;
    size_t off = 0;
    float* h1e     = (float*)(ws + off); off += (size_t)N_EDGES * 64 * 4;
    float* h2e     = (float*)(ws + off); off += (size_t)N_EDGES * 128 * 4;
    float* nodeacc = (float*)(ws + off); off += (size_t)N_NODES * 64 * 4;
    float* cnt     = (float*)(ws + off); off += (size_t)N_NODES * 4;
    float* h1      = (float*)(ws + off); off += (size_t)N_NODES * 64 * 4;
    float* h2      = (float*)(ws + off); off += (size_t)N_NODES * 64 * 4;
    unsigned short* B1p = (unsigned short*)(ws + off); off += (size_t)9   * 2048 * 2;
    unsigned short* B2p = (unsigned short*)(ws + off); off += (size_t)258 * 2048 * 2;

    // prep
    zero_kernel<<<(N_NODES * 64 + 255) / 256, 256, 0, stream>>>(nodeacc, N_NODES * 64);
    zero_kernel<<<(N_NODES + 255) / 256, 256, 0, stream>>>(cnt, N_NODES);
    count_kernel<<<N_EDGES / 256, 256, 0, stream>>>(dst, cnt);
    edge_mlp_kernel<<<N_EDGES, 128, 0, stream>>>(ea, e1w1, e1b1, e2w1, e2b1, h1e, h2e);
    prep_b_kernel<<<(9 * 2048 + 255) / 256, 256, 0, stream>>>(B1p, e1w2, e1b2, 64, 2, 9);
    prep_b_kernel<<<(258 * 2048 + 255) / 256, 256, 0, stream>>>(B2p, e2w2, e2b2, 128, 6, 258);

    // layer 1
    msg1_kernel<<<N_EDGES / 16, 128, 0, stream>>>(x, h1e, src, dst, B1p, nodeacc);
    node_update_kernel<<<N_NODES, 64, 0, stream>>>(nodeacc, cnt, x, 4,
                                                   root1, bias1, ln1g, ln1b, h1);

    // layer 2
    zero_kernel<<<(N_NODES * 64 + 255) / 256, 256, 0, stream>>>(nodeacc, N_NODES * 64);
    msg2_kernel<<<N_EDGES / 16, 128, 0, stream>>>(h1, h2e, src, dst, B2p, nodeacc);
    node_update_kernel<<<N_NODES, 64, 0, stream>>>(nodeacc, cnt, h1, 64,
                                                   root2, bias2, ln2g, ln2b, h2);

    // pool
    pool_kernel<<<64, 256, 0, stream>>>(h2, (float*)d_out);
}
